// Multi_head_attention_71554155151408
// MI455X (gfx1250) — matrix operand order repaired and checked by reading, compiled, not run on hardware
//
#include <hip/hip_runtime.h>
#include <hip/hip_bf16.h>

// ---------------------------------------------------------------------------
// Multi-head attention (B=2, L=2048, D=1024, H=8, head_dim=128) for gfx1250.
// bf16 WMMA GEMMs + flash attention + layernorm.
// Data movement: global_load_async_to_lds_b128 (ASYNCcnt) for non-transposed
// tiles, TDM tensor_load_to_lds (TENSORcnt) for the attention K tile.
// ---------------------------------------------------------------------------

typedef __bf16 bf16;
typedef __attribute__((ext_vector_type(16))) __bf16 v16bf;
typedef __attribute__((ext_vector_type(8)))  __bf16 v8bf;
typedef __attribute__((ext_vector_type(8)))  float  v8f;
typedef __attribute__((ext_vector_type(4)))  unsigned v4u;
typedef __attribute__((ext_vector_type(8)))  int      v8i_;
typedef __attribute__((ext_vector_type(4)))  int      v4i_;

union V16U  { v16bf v; v8bf h[2]; };
union BPack { __bf16 b[2]; unsigned u; };

#define B_  2
#define L_  2048
#define D_  1024
#define H_  8
#define HD_ 128
#define M_  (B_*L_)

// async global->LDS copy of 16 bytes (ASYNCcnt-tracked)
__device__ __forceinline__ void async_b128(unsigned lds_off, const void* gptr) {
    unsigned long long ga = (unsigned long long)gptr;
    asm volatile("global_load_async_to_lds_b128 %0, %1, off"
                 :: "v"(lds_off), "v"(ga) : "memory");
}
__device__ __forceinline__ void wait_async0() {
    asm volatile("s_wait_asynccnt 0x0" ::: "memory");
}

// ---------------------------------------------------------------------------
// fp32 -> bf16 elementwise conversion
// ---------------------------------------------------------------------------
__global__ void cvt_kernel(const float* __restrict__ in, bf16* __restrict__ out, int n) {
    int i = blockIdx.x * blockDim.x + threadIdx.x;
    int stride = gridDim.x * blockDim.x;
    for (; i < n; i += stride) out[i] = (bf16)in[i];
}

// ---------------------------------------------------------------------------
// Tiled bf16 GEMM: Y = X(MxK) @ W(KxN) + bias, M=4096, N=K=1024.
// Block tile 128x128, BK=32.  8 waves, each wave 64x32 (4x2 frags of 16x16).
// mode 0: store bf16, column-permuted to head-major [b][h][l][c], *scale.
// mode 1: store fp32 to outF with fp32 residual added.
// ---------------------------------------------------------------------------
__global__ __launch_bounds__(256) void gemm_kernel(
    const bf16* __restrict__ X, const bf16* __restrict__ W,
    const float* __restrict__ bias,
    bf16* __restrict__ outPerm, float scale,
    float* __restrict__ outF, const float* __restrict__ residual,
    int mode)
{
    const int N = D_, K = D_;
    __shared__ alignas(16) bf16 As[128][40];   // [m][k], padded stride 40 (80B)
    __shared__ alignas(16) bf16 Bs[128][40];   // transposed: [n][k]

    const int tid  = threadIdx.x;
    const int lane = tid & 31;
    const int wid  = tid >> 5;
    const int half = lane >> 4;
    const int ln   = lane & 15;
    const int row0 = blockIdx.y * 128;
    const int n0   = blockIdx.x * 128;
    const int mb   = (wid >> 2) * 64;   // wave M offset
    const int nb   = (wid & 3) * 32;    // wave N offset

    v8f acc[4][2];
    for (int i = 0; i < 4; ++i)
        for (int j = 0; j < 2; ++j)
            for (int r = 0; r < 8; ++r) acc[i][j][r] = 0.0f;

    for (int k0 = 0; k0 < K; k0 += 32) {
        // stage A tile (128x32) via async global->LDS b128 copies
        for (int i = 0; i < 2; ++i) {
            int q  = tid * 2 + i;
            int ar = q >> 2, ac = q & 3;
            async_b128((unsigned)(size_t)&As[ar][ac * 8],
                       &X[(size_t)(row0 + ar) * K + k0 + ac * 8]);
        }
        // stage B tile (32x128) transposed into [n][k]: read two k-rows,
        // pack bf16 pairs, b32 LDS stores (k-pairs contiguous in [n][k])
        {
            int kp = tid >> 4;          // 0..15 -> k = 2kp, 2kp+1
            int bn = tid & 15;          // 8-col chunk
            v8bf w0 = *(const v8bf*)&W[(size_t)(k0 + 2 * kp)     * N + n0 + bn * 8];
            v8bf w1 = *(const v8bf*)&W[(size_t)(k0 + 2 * kp + 1) * N + n0 + bn * 8];
            for (int j = 0; j < 8; ++j) {
                BPack pk; pk.b[0] = w0[j]; pk.b[1] = w1[j];
                *(unsigned*)&Bs[bn * 8 + j][2 * kp] = pk.u;
            }
        }
        wait_async0();
        __syncthreads();

        // fragments per ISA 16-bit A(16x32)/B(32x16) VGPR layouts
        V16U a[4], bm[2];
        for (int i = 0; i < 4; ++i) {
            int m    = mb + i * 16 + ln;
            int koff = half * 8;
            a[i].h[0] = *(const v8bf*)&As[m][koff];
            a[i].h[1] = *(const v8bf*)&As[m][16 + koff];
        }
        for (int j = 0; j < 2; ++j) {
            int n    = nb + j * 16 + ln;
            int koff = half * 8;
            bm[j].h[0] = *(const v8bf*)&Bs[n][koff];
            bm[j].h[1] = *(const v8bf*)&Bs[n][16 + koff];
        }
        for (int i = 0; i < 4; ++i)
            for (int j = 0; j < 2; ++j)
                acc[i][j] = __builtin_amdgcn_wmma_f32_16x16x32_bf16(
                    false, a[i].v, false, bm[j].v, (short)0, acc[i][j], false, false);
        __syncthreads();
    }

    // epilogue
    for (int i = 0; i < 4; ++i) {
        for (int j = 0; j < 2; ++j) {
            int col    = n0 + nb + j * 16 + ln;
            float bval = bias[col];
            for (int r = 0; r < 8; ++r) {
                int row   = row0 + mb + i * 16 + r + half * 8;
                float val = acc[i][j][r] + bval;
                if (mode == 0) {
                    val *= scale;
                    int bb = row >> 11;       // row / L_
                    int l  = row & (L_ - 1);
                    int hh = col & 7;         // faithful strided head split
                    int cc = col >> 3;
                    outPerm[(((size_t)bb * H_ + hh) * L_ + l) * HD_ + cc] = (bf16)val;
                } else {
                    size_t idx = (size_t)row * N + col;
                    outF[idx] = val + residual[idx];
                }
            }
        }
    }
}

// ---------------------------------------------------------------------------
// Flash attention per (b,h).  64 q rows / block, 4 waves x 16 rows.
// Q pre-scaled by 1/sqrt(D).  K tile fetched by the Tensor Data Mover with
// LDS padding (4 DWORDs every 64 DWORDs -> 272B row stride).
// ---------------------------------------------------------------------------
__global__ __launch_bounds__(128) void attn_kernel(
    const bf16* __restrict__ Qh, const bf16* __restrict__ Kh,
    const bf16* __restrict__ Vh,
    const unsigned char* __restrict__ mask,   // bool bytes, (B,Lq,Lk), True=masked
    bf16* __restrict__ Aout)                  // [b][l][h*128+c]
{
    __shared__ alignas(16) bf16 Qs[64][136];
    __shared__ alignas(16) bf16 Ks[32][136];
    __shared__ alignas(16) bf16 Vts[128][40];   // transposed V: [feat][kpos]
    __shared__ alignas(16) bf16 Ps[4][16][40];  // per-wave P transpose staging

    const int tid  = threadIdx.x;
    const int lane = tid & 31;
    const int wid  = tid >> 5;
    const int half = lane >> 4;
    const int ln   = lane & 15;
    const int bh   = blockIdx.y;
    const int b    = bh >> 3;
    const int h    = bh & 7;
    const int q0   = blockIdx.x * 64;
    const int mrow = wid * 16;

    const bf16* Qb = Qh + (size_t)(b * H_ + h) * L_ * HD_;
    const bf16* Kb = Kh + (size_t)(b * H_ + h) * L_ * HD_;
    const bf16* Vb = Vh + (size_t)(b * H_ + h) * L_ * HD_;
    const unsigned char* mbase = mask + (size_t)b * L_ * L_;

    // stage Q tile (64x128) once, async
    for (int i = 0; i < 8; ++i) {
        int q = tid + i * 128;
        int r = q >> 4, c = q & 15;
        async_b128((unsigned)(size_t)&Qs[r][c * 8],
                   &Qb[(size_t)(q0 + r) * HD_ + c * 8]);
    }
    wait_async0();

    v8f O[8];
    for (int jf = 0; jf < 8; ++jf)
        for (int r = 0; r < 8; ++r) O[jf][r] = 0.0f;
    float mprev[8], lsum[8];
    for (int r = 0; r < 8; ++r) { mprev[r] = -3.0e38f; lsum[r] = 0.0f; }

    // invariant parts of the TDM descriptor (D#), per CDNA5 ISA
    const unsigned ks_base = (unsigned)(size_t)&Ks[0][0];
    for (int kb = 0; kb < L_ / 32; ++kb) {
        int kpos0 = kb * 32;

        // --- K tile (32 x 128, natural layout) via TDM, issued by wave 0 ---
        if (wid == 0) {
            unsigned long long ga =
                (unsigned long long)(const void*)(Kb + (size_t)kpos0 * HD_);
            v4u g0;
            g0[0] = 1u;                                   // count=1, user mode
            g0[1] = ks_base;                              // lds_addr
            g0[2] = (unsigned)ga;                         // global_addr[31:0]
            g0[3] = (unsigned)((ga >> 32) & 0x1FFFFFFu)   // global_addr[56:32]
                  | (2u << 30);                           // type=2 ("image")
            v8i_ g1;
            g1[0] = (int)((1u << 16)      // data_size = 2 bytes
                  |       (1u << 20)      // pad_enable
                  |       (5u << 22)      // pad_interval: every 64 DWORDs (256B)
                  |       (3u << 25));    // pad_amount: 4 DWORDs (16B)
            g1[1] = (int)((unsigned)HD_ << 16);           // tensor_dim0 = 128
            g1[2] = (int)(((unsigned)L_ & 0xFFFFu) << 16);// tensor_dim1 = 2048 (lo16)
            g1[3] = (int)((unsigned)HD_ << 16);           // tile_dim0 = 128
            g1[4] = (int)32u;                             // tile_dim1 = 32, tile_dim2=0
            g1[5] = (int)HD_;                             // tensor_dim0_stride = 128
            g1[6] = 0;
            g1[7] = 0;
            v4i_ gz  = {0, 0, 0, 0};
            v8i_ gz8 = {0, 0, 0, 0, 0, 0, 0, 0};
            __builtin_amdgcn_tensor_load_to_lds(g0, g1, gz, gz, gz8, 0);
        }

        // --- V tile transposed ([feat][kpos]) with packed b32 stores ---
        for (int i = 0; i < 2; ++i) {
            int q  = tid + i * 128;
            int kp = q >> 4;            // 0..15 -> kpos pair 2kp, 2kp+1
            int c  = q & 15;
            v8bf w0 = *(const v8bf*)&Vb[(size_t)(kpos0 + 2 * kp)     * HD_ + c * 8];
            v8bf w1 = *(const v8bf*)&Vb[(size_t)(kpos0 + 2 * kp + 1) * HD_ + c * 8];
            for (int j = 0; j < 8; ++j) {
                BPack pk; pk.b[0] = w0[j]; pk.b[1] = w1[j];
                *(unsigned*)&Vts[c * 8 + j][2 * kp] = pk.u;
            }
        }
        if (wid == 0) __builtin_amdgcn_s_wait_tensorcnt(0);
        __syncthreads();

        // S = Q @ K^T  (16 rows x 32 kpos), 2 n-frags, 4 k-steps each
        v8f s[2];
        for (int nf = 0; nf < 2; ++nf) {
            for (int r = 0; r < 8; ++r) s[nf][r] = 0.0f;
            for (int kk = 0; kk < 4; ++kk) {
                V16U a, bm;
                int m    = mrow + ln;
                int koff = half * 8;
                a.h[0] = *(const v8bf*)&Qs[m][kk * 32 + koff];
                a.h[1] = *(const v8bf*)&Qs[m][kk * 32 + 16 + koff];
                int np = nf * 16 + ln;
                bm.h[0] = *(const v8bf*)&Ks[np][kk * 32 + koff];
                bm.h[1] = *(const v8bf*)&Ks[np][kk * 32 + 16 + koff];
                s[nf] = __builtin_amdgcn_wmma_f32_16x16x32_bf16(
                    false, a.v, false, bm.v, (short)0, s[nf], false, false);
            }
            // mask (True -> -1e10), applied to the already-scaled scores
            for (int r = 0; r < 8; ++r) {
                int qrow = q0 + mrow + r + half * 8;
                int kcol = kpos0 + nf * 16 + ln;
                if (mbase[(size_t)qrow * L_ + kcol]) s[nf][r] = -1.0e10f;
            }
        }

        // row max across the 16-lane column group
        float mcur[8];
        for (int r = 0; r < 8; ++r) mcur[r] = fmaxf(s[0][r], s[1][r]);
        for (int off = 8; off >= 1; off >>= 1)
            for (int r = 0; r < 8; ++r)
                mcur[r] = fmaxf(mcur[r], __shfl_xor(mcur[r], off, 16));

        // online softmax update
        float rs[8];
        for (int r = 0; r < 8; ++r) {
            float mnew = fmaxf(mprev[r], mcur[r]);
            float corr = __expf(mprev[r] - mnew);
            s[0][r] = __expf(s[0][r] - mnew);
            s[1][r] = __expf(s[1][r] - mnew);
            rs[r]   = s[0][r] + s[1][r];
            mprev[r] = mnew;
            lsum[r] *= corr;
            for (int jf = 0; jf < 8; ++jf) O[jf][r] *= corr;
        }
        for (int off = 8; off >= 1; off >>= 1)
            for (int r = 0; r < 8; ++r) rs[r] += __shfl_xor(rs[r], off, 16);
        for (int r = 0; r < 8; ++r) lsum[r] += rs[r];

        // transpose P (C-layout -> A-layout) through per-wave LDS
        for (int nf = 0; nf < 2; ++nf)
            for (int r = 0; r < 8; ++r)
                Ps[wid][r + half * 8][nf * 16 + ln] = (bf16)s[nf][r];

        // O += P(16x32) @ V(32x128)
        V16U pa;
        pa.h[0] = *(const v8bf*)&Ps[wid][ln][half * 8];
        pa.h[1] = *(const v8bf*)&Ps[wid][ln][16 + half * 8];
        for (int jf = 0; jf < 8; ++jf) {
            V16U vb;
            vb.h[0] = *(const v8bf*)&Vts[jf * 16 + ln][half * 8];
            vb.h[1] = *(const v8bf*)&Vts[jf * 16 + ln][16 + half * 8];
            O[jf] = __builtin_amdgcn_wmma_f32_16x16x32_bf16(
                false, pa.v, false, vb.v, (short)0, O[jf], false, false);
        }
        __syncthreads();
    }

    // normalize and store concat layout: Aout[b][l][h*128 + c]
    for (int r = 0; r < 8; ++r) {
        float inv = 1.0f / lsum[r];
        int qrow  = q0 + mrow + r + half * 8;
        for (int jf = 0; jf < 8; ++jf) {
            int col = h * HD_ + jf * 16 + ln;
            Aout[(size_t)(b * L_ + qrow) * D_ + col] = (bf16)(O[jf][r] * inv);
        }
    }
}

// ---------------------------------------------------------------------------
// LayerNorm (unbiased var, ddof=1, eps=1e-8), in-place, one block per row.
// ---------------------------------------------------------------------------
__global__ __launch_bounds__(256) void ln_kernel(
    float* __restrict__ out,
    const float* __restrict__ gamma, const float* __restrict__ beta)
{
    __shared__ float red[256];
    const int row = blockIdx.x;
    const int tid = threadIdx.x;
    float* p = out + (size_t)row * D_;

    float x[4];
    float s = 0.0f;
    for (int i = 0; i < 4; ++i) { x[i] = p[tid + i * 256]; s += x[i]; }
    red[tid] = s; __syncthreads();
    for (int off = 128; off >= 1; off >>= 1) {
        if (tid < off) red[tid] += red[tid + off];
        __syncthreads();
    }
    float mu = red[0] / (float)D_;
    __syncthreads();

    float v = 0.0f;
    for (int i = 0; i < 4; ++i) { float d = x[i] - mu; v += d * d; }
    red[tid] = v; __syncthreads();
    for (int off = 128; off >= 1; off >>= 1) {
        if (tid < off) red[tid] += red[tid + off];
        __syncthreads();
    }
    float var = red[0] / (float)(D_ - 1);
    float inv = rsqrtf(var + 1e-8f);
    float g = gamma[0], be = beta[0];
    for (int i = 0; i < 4; ++i)
        p[tid + i * 256] = g * (x[i] - mu) * inv + be;
}

// ---------------------------------------------------------------------------
// Launch
// ---------------------------------------------------------------------------
extern "C" void kernel_launch(void* const* d_in, const int* in_sizes, int n_in,
                              void* d_out, int out_size, void* d_ws, size_t ws_size,
                              hipStream_t stream) {
    (void)in_sizes; (void)n_in; (void)out_size; (void)ws_size;
    const float* en    = (const float*)d_in[0];
    const float* de    = (const float*)d_in[1];
    const unsigned char* mask = (const unsigned char*)d_in[2];  // bool array
    const float* Wq    = (const float*)d_in[3];
    const float* bq    = (const float*)d_in[4];
    const float* Wk    = (const float*)d_in[5];
    const float* bk    = (const float*)d_in[6];
    const float* Wv    = (const float*)d_in[7];
    const float* bv    = (const float*)d_in[8];
    const float* Wo    = (const float*)d_in[9];
    const float* bo    = (const float*)d_in[10];
    const float* gamma = (const float*)d_in[11];
    const float* beta  = (const float*)d_in[12];
    float* out = (float*)d_out;

    char* p = (char*)d_ws;
    auto carve = [&](size_t bytes) {
        char* r = p;
        p += (bytes + 255) & ~(size_t)255;
        return r;
    };
    bf16* deb = (bf16*)carve((size_t)M_ * D_ * 2);
    bf16* enb = (bf16*)carve((size_t)M_ * D_ * 2);
    bf16* wqb = (bf16*)carve((size_t)D_ * D_ * 2);
    bf16* wkb = (bf16*)carve((size_t)D_ * D_ * 2);
    bf16* wvb = (bf16*)carve((size_t)D_ * D_ * 2);
    bf16* wob = (bf16*)carve((size_t)D_ * D_ * 2);
    bf16* Qh  = (bf16*)carve((size_t)M_ * D_ * 2);   // [b][h][l][c], pre-scaled
    bf16* Kh  = (bf16*)carve((size_t)M_ * D_ * 2);
    bf16* Vh  = (bf16*)carve((size_t)M_ * D_ * 2);
    bf16* Ab  = (bf16*)carve((size_t)M_ * D_ * 2);   // concat attention output

    cvt_kernel<<<512, 256, 0, stream>>>(de, deb, M_ * D_);
    cvt_kernel<<<512, 256, 0, stream>>>(en, enb, M_ * D_);
    cvt_kernel<<<256, 256, 0, stream>>>(Wq, wqb, D_ * D_);
    cvt_kernel<<<256, 256, 0, stream>>>(Wk, wkb, D_ * D_);
    cvt_kernel<<<256, 256, 0, stream>>>(Wv, wvb, D_ * D_);
    cvt_kernel<<<256, 256, 0, stream>>>(Wo, wob, D_ * D_);

    dim3 gg(D_ / 128, M_ / 128);
    // Q gets 1/sqrt(d_model) = 1/32 folded in
    gemm_kernel<<<gg, 256, 0, stream>>>(deb, wqb, bq, Qh, 0.03125f, nullptr, nullptr, 0);
    gemm_kernel<<<gg, 256, 0, stream>>>(enb, wkb, bk, Kh, 1.0f,     nullptr, nullptr, 0);
    gemm_kernel<<<gg, 256, 0, stream>>>(enb, wvb, bv, Vh, 1.0f,     nullptr, nullptr, 0);

    dim3 ga(L_ / 64, B_ * H_);
    attn_kernel<<<ga, 128, 0, stream>>>(Qh, Kh, Vh, mask, Ab);

    // result = A @ Wo + bo + de  (fp32)
    gemm_kernel<<<gg, 256, 0, stream>>>(Ab, wob, bo, nullptr, 1.0f, out, de, 1);

    ln_kernel<<<M_, 256, 0, stream>>>(out, gamma, beta);
}
